// qcGEM_18854906429825
// MI455X (gfx1250) — matrix operand order repaired and checked by reading, not compiled, not run on hardware
//
#include <hip/hip_runtime.h>
#include <hip/hip_bf16.h>

typedef __attribute__((ext_vector_type(16))) _Float16 v16h;
typedef __attribute__((ext_vector_type(8)))  _Float16 v8h;
typedef __attribute__((ext_vector_type(8)))  float    v8f;

#define NN   10000
#define EE   320000
#define DD   128
#define EDI  64
#define EHD  256   // edge hidden
#define EOD  128   // edge out
#define NHD  256   // node hidden
#define NOD  128   // node out
#define K1   450   // 3*DD + 2 + EDI
#define K1P  480   // padded to multiple of 32
#define K3   256   // DD + EOD
#define EM   64    // edges per workgroup (4 M-tiles)

__device__ __forceinline__ float gelu_exact(float x) {
    return 0.5f * x * (1.0f + erff(x * 0.7071067811865476f));
}

__device__ __forceinline__ unsigned pack2(float a, float b) {
    union { _Float16 h[2]; unsigned u; } x;
    x.h[0] = (_Float16)a; x.h[1] = (_Float16)b;
    return x.u;
}

// A-fragment for v_wmma_f32_16x16x32_f16:
// lane L holds row m = L%16; its 16 f16 are K = kt*32 + 16*(L/16) + {0..15}
__device__ __forceinline__ v16h load_a_frag(const _Float16* row, int kt, int half) {
    const _Float16* p = row + kt * 32 + half * 16;
    union { v16h v; v8h h[2]; } u;
    u.h[0] = *(const v8h*)(p);
    u.h[1] = *(const v8h*)(p + 8);
    return u.v;
}

__global__ __launch_bounds__(256) void zero_kernel(float* __restrict__ p, int n) {
    int i = blockIdx.x * 256 + threadIdx.x;
    if (i < n) p[i] = 0.0f;
}

// Pack all 4 weight matrices into f16, transposed to [Ncols][K] so a lane's
// B-fragment (col = lane%16, K chunk = 16*(lane/16)) is one contiguous 32B load.
__global__ __launch_bounds__(256) void prep_kernel(
    const float* __restrict__ eW1, const float* __restrict__ eW2,
    const float* __restrict__ nW1, const float* __restrict__ nW2,
    _Float16* __restrict__ WT1, _Float16* __restrict__ WT2,
    _Float16* __restrict__ WT3, _Float16* __restrict__ WT4)
{
    int i = blockIdx.x * 256 + threadIdx.x;
    if (i < EHD * K1P) { int n = i / K1P, k = i % K1P;
        WT1[i] = (k < K1) ? (_Float16)eW1[k * EHD + n] : (_Float16)0.0f; }
    if (i < EOD * EHD) { int n = i / EHD, k = i % EHD;
        WT2[i] = (_Float16)eW2[k * EOD + n]; }
    if (i < NHD * K3)  { int n = i / K3,  k = i % K3;
        WT3[i] = (_Float16)nW1[k * NHD + n]; }
    if (i < NOD * NHD) { int n = i / NHD, k = i % NHD;
        WT4[i] = (_Float16)nW2[k * NOD + n]; }
}

// 64 edges / workgroup, 8 waves. Wave w owns N-tiles {2w, 2w+1} across all
// 4 M-tiles -> each B fragment feeds 4 WMMAs (4x less weight traffic).
__global__ __launch_bounds__(256) void edge_kernel(
    const float* __restrict__ nf, const float* __restrict__ ef,
    const long long* __restrict__ eidx,
    const _Float16* __restrict__ WT1, const _Float16* __restrict__ WT2,
    const float* __restrict__ eb1, const float* __restrict__ eg1, const float* __restrict__ ebt1,
    const float* __restrict__ eb2, const float* __restrict__ eg2, const float* __restrict__ ebt2,
    float* __restrict__ edge_out, float* __restrict__ agg)
{
    // 127KB LDS: CDNA5 WGP has 320KB (160KB/CU-half in CUMODE) -> 2 blocks/WGP
    __shared__ __align__(32) char smem[EM * K1P * 2 + EM * EHD * 4];
    _Float16 (*sA)[K1P]  = (_Float16(*)[K1P])smem;                 // concat rows f16
    float    (*sOut)[EHD] = (float(*)[EHD])(smem + EM * K1P * 2);  // f32 GEMM out
    _Float16 (*sA2)[EHD] = (_Float16(*)[EHD])smem;                 // overlays dead sA
    __shared__ float sR1[EM][4], sR2[EM][4], sMean[EM], sRstd[EM];
    __shared__ int   sCol[EM];

    const int tid = threadIdx.x;
    const int el  = tid >> 2;          // edge within tile (0..63)
    const int p   = tid & 3;           // 4-way partition per edge
    const long long eg = (long long)blockIdx.x * EM + el;

    // ---- Stage 1: gather src/tgt (L2-resident), build concat rows in LDS ----
    {
        long long r = eidx[eg];
        long long c = eidx[(long long)EE + eg];
        if (p == 0) sCol[el] = (int)c;
        const float4* sp4 = (const float4*)(nf + r * DD + p * 32);
        const float4* tp4 = (const float4*)(nf + c * DD + p * 32);
        float loc = 0.0f;
        #pragma unroll
        for (int j4 = 0; j4 < 8; ++j4) {
            float4 s = sp4[j4], t = tp4[j4];
            float d0 = s.x - t.x, d1 = s.y - t.y, d2 = s.z - t.z, d3 = s.w - t.w;
            int d = p * 32 + j4 * 4;
            *(unsigned*)&sA[el][d]              = pack2(s.x, s.y);
            *(unsigned*)&sA[el][d + 2]          = pack2(s.z, s.w);
            *(unsigned*)&sA[el][DD + d]         = pack2(t.x, t.y);
            *(unsigned*)&sA[el][DD + d + 2]     = pack2(t.z, t.w);
            *(unsigned*)&sA[el][2 * DD + d]     = pack2(d0, d1);
            *(unsigned*)&sA[el][2 * DD + d + 2] = pack2(d2, d3);
            loc += d0 * d0 + d1 * d1 + d2 * d2 + d3 * d3;
        }
        sR1[el][p] = loc;
        const float4* ef4 = (const float4*)(ef + eg * EDI + p * 16);
        #pragma unroll
        for (int j4 = 0; j4 < 4; ++j4) {
            float4 e = ef4[j4];
            int q = 3 * DD + 2 + p * 16 + j4 * 4;
            *(unsigned*)&sA[el][q]     = pack2(e.x, e.y);
            *(unsigned*)&sA[el][q + 2] = pack2(e.z, e.w);
        }
        for (int i = p; i < 15; i += 4)                 // zero pad 450..479
            *(unsigned*)&sA[el][K1 + 2 * i] = 0u;
    }
    __syncthreads();
    if (p == 0) {
        float rad = sR1[el][0] + sR1[el][1] + sR1[el][2] + sR1[el][3];
        *(unsigned*)&sA[el][3 * DD] = pack2(rad, sqrtf(rad + 1e-8f));
    }
    __syncthreads();

    const int lane = tid & 31, wave = tid >> 5;
    const int m = lane & 15, half = lane >> 4;

    // ---- GEMM1: [64 x 480] @ [480 x 256] -> sOut ----
    {
        v8f acc[4][2] = {};
        for (int kt = 0; kt < K1P / 32; ++kt) {
            v16h a[4];
            #pragma unroll
            for (int mt = 0; mt < 4; ++mt)
                a[mt] = load_a_frag(&sA[mt * 16 + m][0], kt, half);
            v16h b[2];
            #pragma unroll
            for (int t = 0; t < 2; ++t) {
                int n = (wave * 2 + t) * 16 + m;
                b[t] = *(const v16h*)(WT1 + (long long)n * K1P + kt * 32 + half * 16);
            }
            #pragma unroll
            for (int mt = 0; mt < 4; ++mt)
                #pragma unroll
                for (int t = 0; t < 2; ++t)
                    acc[mt][t] = __builtin_amdgcn_wmma_f32_16x16x32_f16(
                        false, a[mt], false, b[t], (short)0, acc[mt][t], false, false);
        }
        #pragma unroll
        for (int mt = 0; mt < 4; ++mt)
            #pragma unroll
            for (int t = 0; t < 2; ++t) {
                int nc = (wave * 2 + t) * 16 + m;
                #pragma unroll
                for (int r = 0; r < 8; ++r)
                    sOut[mt * 16 + half * 8 + r][nc] = acc[mt][t][r];
            }
    }
    __syncthreads();

    // ---- bias + LayerNorm + GELU over 256 -> sA2 (f16, overlays sA) ----
    {
        float s1 = 0.0f, s2 = 0.0f;
        #pragma unroll
        for (int j = 0; j < 64; ++j) {
            int c = p * 64 + j;
            float x = sOut[el][c] + eb1[c];
            sOut[el][c] = x;
            s1 += x; s2 += x * x;
        }
        sR1[el][p] = s1; sR2[el][p] = s2;
        __syncthreads();
        if (p == 0) {
            float mm = sR1[el][0] + sR1[el][1] + sR1[el][2] + sR1[el][3];
            float vv = sR2[el][0] + sR2[el][1] + sR2[el][2] + sR2[el][3];
            mm *= (1.0f / EHD);
            vv = vv * (1.0f / EHD) - mm * mm;
            sMean[el] = mm; sRstd[el] = rsqrtf(vv + 1e-5f);
        }
        __syncthreads();
        float mm = sMean[el], rs = sRstd[el];
        #pragma unroll
        for (int j = 0; j < 64; j += 2) {
            int c = p * 64 + j;
            float y0 = (sOut[el][c]     - mm) * rs * eg1[c]     + ebt1[c];
            float y1 = (sOut[el][c + 1] - mm) * rs * eg1[c + 1] + ebt1[c + 1];
            *(unsigned*)&sA2[el][c] = pack2(gelu_exact(y0), gelu_exact(y1));
        }
    }
    __syncthreads();

    // ---- GEMM2: [64 x 256] @ [256 x 128] -> sOut[:, :128] ----
    // wave w owns N-tile w; B fragment reused across 4 M-tiles.
    {
        v8f acc[4] = {};
        for (int kt = 0; kt < EHD / 32; ++kt) {
            int n = wave * 16 + m;
            v16h b = *(const v16h*)(WT2 + n * EHD + kt * 32 + half * 16);
            #pragma unroll
            for (int mt = 0; mt < 4; ++mt) {
                v16h a = load_a_frag(&sA2[mt * 16 + m][0], kt, half);
                acc[mt] = __builtin_amdgcn_wmma_f32_16x16x32_f16(
                    false, a, false, b, (short)0, acc[mt], false, false);
            }
        }
        __syncthreads();
        #pragma unroll
        for (int mt = 0; mt < 4; ++mt) {
            int nc = wave * 16 + m;
            #pragma unroll
            for (int r = 0; r < 8; ++r)
                sOut[mt * 16 + half * 8 + r][nc] = acc[mt][r];
        }
    }
    __syncthreads();

    // ---- bias + LN + GELU over 128, write edge_out + scatter-add agg ----
    {
        float s1 = 0.0f, s2 = 0.0f;
        #pragma unroll
        for (int j = 0; j < 32; ++j) {
            int c = p * 32 + j;
            float x = sOut[el][c] + eb2[c];
            sOut[el][c] = x;
            s1 += x; s2 += x * x;
        }
        sR1[el][p] = s1; sR2[el][p] = s2;
        __syncthreads();
        if (p == 0) {
            float mm = sR1[el][0] + sR1[el][1] + sR1[el][2] + sR1[el][3];
            float vv = sR2[el][0] + sR2[el][1] + sR2[el][2] + sR2[el][3];
            mm *= (1.0f / EOD);
            vv = vv * (1.0f / EOD) - mm * mm;
            sMean[el] = mm; sRstd[el] = rsqrtf(vv + 1e-5f);
        }
        __syncthreads();
        float mm = sMean[el], rs = sRstd[el];
        int ct = sCol[el];
        #pragma unroll
        for (int j = 0; j < 32; ++j) {
            int c = p * 32 + j;
            float y = (sOut[el][c] - mm) * rs * eg2[c] + ebt2[c];
            float g = gelu_exact(y);
            edge_out[eg * EOD + c] = g;
            atomicAdd(&agg[(long long)ct * EOD + c], g);  // L2-resident scatter
        }
    }
}

__global__ __launch_bounds__(128) void node_kernel(
    const float* __restrict__ nf, const float* __restrict__ agg,
    const _Float16* __restrict__ WT3, const _Float16* __restrict__ WT4,
    const float* __restrict__ nb1, const float* __restrict__ ng1, const float* __restrict__ nbt1,
    const float* __restrict__ nb2, const float* __restrict__ ng2, const float* __restrict__ nbt2,
    float* __restrict__ node_out)
{
    __shared__ __align__(16) _Float16 sA2[16][NHD];
    __shared__ __align__(16) float    sOut[16][NHD];
    __shared__ float sR1[16][8], sR2[16][8], sMean[16], sRstd[16];

    const int tid = threadIdx.x, el = tid >> 3, p = tid & 7;
    const long long ng = (long long)blockIdx.x * 16 + el;

    // build [node_features | agg] row as f16
    #pragma unroll
    for (int j = 0; j < 32; j += 2) {
        int d = p * 32 + j;
        float v0 = (d < DD)     ? nf[ng * DD + d]           : agg[ng * EOD + (d - DD)];
        float v1 = (d + 1 < DD) ? nf[ng * DD + d + 1]       : agg[ng * EOD + (d + 1 - DD)];
        *(unsigned*)&sA2[el][d] = pack2(v0, v1);
    }
    __syncthreads();

    const int lane = tid & 31, wave = tid >> 5;
    const int m = lane & 15, half = lane >> 4;

    // GEMM1: [16 x 256] @ [256 x 256]
    {
        v8f acc[4] = {};
        for (int kt = 0; kt < K3 / 32; ++kt) {
            v16h a = load_a_frag(&sA2[m][0], kt, half);
            #pragma unroll
            for (int t = 0; t < 4; ++t) {
                int n = (wave * 4 + t) * 16 + m;
                v16h b = *(const v16h*)(WT3 + n * K3 + kt * 32 + half * 16);
                acc[t] = __builtin_amdgcn_wmma_f32_16x16x32_f16(
                    false, a, false, b, (short)0, acc[t], false, false);
            }
        }
        #pragma unroll
        for (int t = 0; t < 4; ++t) {
            int nc = (wave * 4 + t) * 16 + m;
            #pragma unroll
            for (int r = 0; r < 8; ++r) sOut[half * 8 + r][nc] = acc[t][r];
        }
    }
    __syncthreads();

    // bias + LN + GELU over 256 -> sA2
    {
        float s1 = 0.0f, s2 = 0.0f;
        #pragma unroll
        for (int j = 0; j < 32; ++j) {
            int c = p * 32 + j;
            float x = sOut[el][c] + nb1[c];
            sOut[el][c] = x;
            s1 += x; s2 += x * x;
        }
        sR1[el][p] = s1; sR2[el][p] = s2;
        __syncthreads();
        if (p == 0) {
            float mm = 0.0f, vv = 0.0f;
            #pragma unroll
            for (int j = 0; j < 8; ++j) { mm += sR1[el][j]; vv += sR2[el][j]; }
            mm *= (1.0f / NHD);
            vv = vv * (1.0f / NHD) - mm * mm;
            sMean[el] = mm; sRstd[el] = rsqrtf(vv + 1e-5f);
        }
        __syncthreads();
        float mm = sMean[el], rs = sRstd[el];
        #pragma unroll
        for (int j = 0; j < 32; j += 2) {
            int c = p * 32 + j;
            float y0 = (sOut[el][c]     - mm) * rs * ng1[c]     + nbt1[c];
            float y1 = (sOut[el][c + 1] - mm) * rs * ng1[c + 1] + nbt1[c + 1];
            *(unsigned*)&sA2[el][c] = pack2(gelu_exact(y0), gelu_exact(y1));
        }
    }
    __syncthreads();

    // GEMM2: [16 x 256] @ [256 x 128]
    {
        v8f acc[2] = {};
        for (int kt = 0; kt < NHD / 32; ++kt) {
            v16h a = load_a_frag(&sA2[m][0], kt, half);
            #pragma unroll
            for (int t = 0; t < 2; ++t) {
                int n = (wave * 2 + t) * 16 + m;
                v16h b = *(const v16h*)(WT4 + n * NHD + kt * 32 + half * 16);
                acc[t] = __builtin_amdgcn_wmma_f32_16x16x32_f16(
                    false, a, false, b, (short)0, acc[t], false, false);
            }
        }
        __syncthreads();
        #pragma unroll
        for (int t = 0; t < 2; ++t) {
            int nc = (wave * 2 + t) * 16 + m;
            #pragma unroll
            for (int r = 0; r < 8; ++r) sOut[half * 8 + r][nc] = acc[t][r];
        }
    }
    __syncthreads();

    // bias + LN + GELU over 128 -> node_out
    {
        float s1 = 0.0f, s2 = 0.0f;
        #pragma unroll
        for (int j = 0; j < 16; ++j) {
            int c = p * 16 + j;
            float x = sOut[el][c] + nb2[c];
            sOut[el][c] = x;
            s1 += x; s2 += x * x;
        }
        sR1[el][p] = s1; sR2[el][p] = s2;
        __syncthreads();
        if (p == 0) {
            float mm = 0.0f, vv = 0.0f;
            #pragma unroll
            for (int j = 0; j < 8; ++j) { mm += sR1[el][j]; vv += sR2[el][j]; }
            mm *= (1.0f / NOD);
            vv = vv * (1.0f / NOD) - mm * mm;
            sMean[el] = mm; sRstd[el] = rsqrtf(vv + 1e-5f);
        }
        __syncthreads();
        float mm = sMean[el], rs = sRstd[el];
        #pragma unroll
        for (int j = 0; j < 16; ++j) {
            int c = p * 16 + j;
            float y = (sOut[el][c] - mm) * rs * ng2[c] + nbt2[c];
            node_out[ng * NOD + c] = gelu_exact(y);
        }
    }
}

extern "C" void kernel_launch(void* const* d_in, const int* in_sizes, int n_in,
                              void* d_out, int out_size, void* d_ws, size_t ws_size,
                              hipStream_t stream)
{
    (void)in_sizes; (void)n_in; (void)out_size; (void)ws_size;
    const float*     nf   = (const float*)d_in[0];
    const float*     ef   = (const float*)d_in[1];
    const long long* eidx = (const long long*)d_in[2];  // jnp.int64
    const float* eW1 = (const float*)d_in[3];
    const float* eb1 = (const float*)d_in[4];
    const float* eg1 = (const float*)d_in[5];
    const float* ebt1= (const float*)d_in[6];
    const float* eW2 = (const float*)d_in[7];
    const float* eb2 = (const float*)d_in[8];
    const float* eg2 = (const float*)d_in[9];
    const float* ebt2= (const float*)d_in[10];
    const float* nW1 = (const float*)d_in[11];
    const float* nb1 = (const float*)d_in[12];
    const float* ng1 = (const float*)d_in[13];
    const float* nbt1= (const float*)d_in[14];
    const float* nW2 = (const float*)d_in[15];
    const float* nb2 = (const float*)d_in[16];
    const float* ng2 = (const float*)d_in[17];
    const float* nbt2= (const float*)d_in[18];

    float* node_out = (float*)d_out;
    float* edge_out = node_out + (long long)NN * NOD;

    char* ws = (char*)d_ws;
    float* agg = (float*)ws;                              // N*EO f32 = 5.12 MB
    size_t off = (size_t)NN * EOD * sizeof(float);
    _Float16* WT1 = (_Float16*)(ws + off); off += (size_t)EHD * K1P * 2;
    _Float16* WT2 = (_Float16*)(ws + off); off += (size_t)EOD * EHD * 2;
    _Float16* WT3 = (_Float16*)(ws + off); off += (size_t)NHD * K3  * 2;
    _Float16* WT4 = (_Float16*)(ws + off); off += (size_t)NOD * NHD * 2;

    zero_kernel<<<(NN * EOD + 255) / 256, 256, 0, stream>>>(agg, NN * EOD);
    prep_kernel<<<(EHD * K1P + 255) / 256, 256, 0, stream>>>(
        eW1, eW2, nW1, nW2, WT1, WT2, WT3, WT4);
    edge_kernel<<<EE / EM, 256, 0, stream>>>(
        nf, ef, eidx, WT1, WT2, eb1, eg1, ebt1, eb2, eg2, ebt2, edge_out, agg);
    node_kernel<<<NN / 16, 128, 0, stream>>>(
        nf, agg, WT3, WT4, nb1, ng1, nbt1, nb2, ng2, nbt2, node_out);
}
